// GNN_85856396247547
// MI455X (gfx1250) — matrix operand order repaired and checked by reading, not compiled, not run on hardware
//
#include <hip/hip_runtime.h>
#include <hip/hip_bf16.h>

typedef __attribute__((ext_vector_type(16))) _Float16 v16h;
typedef __attribute__((ext_vector_type(8)))  float    v8f;
typedef _Float16 half_t;
typedef int v4i_vs __attribute__((vector_size(16)));   // type expected by async-LDS builtin

#define DIMD 128

#if defined(__HIP_DEVICE_COMPILE__) && __has_builtin(__builtin_amdgcn_global_load_async_to_lds_b128)
#define HAVE_ASYNC_LDS 1
#else
#define HAVE_ASYNC_LDS 0
#endif

union Frag16 {
    v16h v;
    _Float16 h[16];
    unsigned u[8];
    float4 f4[2];
};

// ---- A-fragment K map: 16-bit A 16x32 (ISA 7.12.2) ----
// VGPR p (pair of halves): p<4 -> K = 2p + laneHi*8 ; p>=4 -> K = 16 + 2(p-4) + laneHi*8
__device__ __forceinline__ int kmapA(int p, int laneHi) {
    int b = (p < 4) ? (2 * p) : (16 + 2 * (p - 4));
    return b + (laneHi ? 8 : 0);
}

// ======================= WMMA GEMM: out = f(A) @ W + bias ==========================
// A: nrows x 128 f32.  Wt: 128(N) x 128(K) f16 (row n = column n of W).
// BN!=0: A-element transform relu(a*scl[k]+shf[k]) (fused BN+ReLU on input).
// RELU!=0: ReLU on output. Both are compile-time so the inner loops are branch-free.
template <int BN, int RELU>
__global__ void __launch_bounds__(256)
gin_gemm(const float* __restrict__ A, const half_t* __restrict__ Wt,
         const float* __restrict__ bias, float* __restrict__ out, int nrows,
         const float* __restrict__ scl, const float* __restrict__ shf)
{
    __shared__ half_t Bs[DIMD * DIMD];   // 32 KB: whole weight matrix, staged once per block

    // ---- stage Wt -> LDS (async global->LDS when available) ----
#if HAVE_ASYNC_LDS
    for (int i = threadIdx.x; i < DIMD * DIMD / 8; i += 256) {
        __builtin_amdgcn_global_load_async_to_lds_b128(
            (v4i_vs*)(Wt + (size_t)i * 8), (v4i_vs*)(Bs + i * 8), 0, 0);
    }
#if __has_builtin(__builtin_amdgcn_s_wait_asynccnt)
    __builtin_amdgcn_s_wait_asynccnt(0);
#else
    asm volatile("s_wait_asynccnt 0x0" ::: "memory");
#endif
#else
    for (int i = threadIdx.x; i < DIMD * DIMD / 8; i += 256) {
        *(float4*)&Bs[i * 8] = *(const float4*)&Wt[(size_t)i * 8];
    }
#endif
    __syncthreads();

    const int lane   = threadIdx.x & 31;
    const int wave   = threadIdx.x >> 5;
    const int m      = lane & 15;
    const int laneHi = lane >> 4;
    const int rowbase = blockIdx.x * 128 + wave * 16;

    int arow = rowbase + m;
    if (arow >= nrows) arow = nrows - 1;          // clamp: EXEC stays all-ones for WMMA
    const float* arow_p = A + (size_t)arow * DIMD;

    v8f c[8];
#pragma unroll
    for (int n = 0; n < 8; ++n) c[n] = (v8f){0.f, 0.f, 0.f, 0.f, 0.f, 0.f, 0.f, 0.f};

#pragma unroll
    for (int k0 = 0; k0 < DIMD; k0 += 32) {
        if (k0 + 32 < DIMD) __builtin_prefetch(arow_p + k0 + 32, 0, 1);

        Frag16 a;
#pragma unroll
        for (int p = 0; p < 8; ++p) {
            int k = k0 + kmapA(p, laneHi);
            float2 v2 = *(const float2*)(arow_p + k);
            if (BN) {   // compile-time: folds away entirely when BN==0
                v2.x = fmaxf(v2.x * scl[k]     + shf[k],     0.f);
                v2.y = fmaxf(v2.y * scl[k + 1] + shf[k + 1], 0.f);
            }
            a.h[2 * p]     = (_Float16)v2.x;
            a.h[2 * p + 1] = (_Float16)v2.y;
        }

        // B fragment: elements 0-7 hold K=k0+laneHi*8..+7, elements 8-15 hold K=k0+16+laneHi*8..+7
        // -> two 16-byte runs in the LDS-resident row: two ds_load_b128.
#pragma unroll
        for (int n = 0; n < 8; ++n) {
            Frag16 b;
            const half_t* bp = &Bs[(n * 16 + m) * DIMD + k0 + laneHi * 8];
            b.f4[0] = *(const float4*)bp;
            b.f4[1] = *(const float4*)(bp + 16);
            c[n] = __builtin_amdgcn_wmma_f32_16x16x32_f16(
                       false, a.v, false, b.v, (short)0, c[n], false, false);
        }
    }

    // Epilogue: C element r of lane = (M = r + laneHi*8, N = (lane&15) + ntile*16)
#pragma unroll
    for (int n = 0; n < 8; ++n) {
        int col = n * 16 + m;
        float bv = bias[col];
#pragma unroll
        for (int r = 0; r < 8; ++r) {
            int row = rowbase + r + laneHi * 8;
            if (row < nrows) {
                float v = c[n][r] + bv;
                if (RELU) v = fmaxf(v, 0.f);
                out[(size_t)row * DIMD + col] = v;
            }
        }
    }
}

// ======================= helpers ==========================
__global__ void copy_f4(const float4* __restrict__ src, float4* __restrict__ dst, long n4) {
    long i = (long)blockIdx.x * blockDim.x + threadIdx.x;
    if (i < n4) dst[i] = src[i];
}

__global__ void fill_zero(float* p, int n) {
    int i = blockIdx.x * blockDim.x + threadIdx.x;
    if (i < n) p[i] = 0.f;
}

// 6 matrices (3 layers x {lin1,lin2}); wt[(l*2+which)*16384 + n*128 + k] = W[l][k][n] as f16
__global__ void convert_weights(const float* __restrict__ lin1_w,
                                const float* __restrict__ lin2_w, half_t* __restrict__ wt) {
    int gid = blockIdx.x * blockDim.x + threadIdx.x;
    if (gid >= 6 * DIMD * DIMD) return;
    int mat = gid >> 14;
    int idx = gid & 16383;
    int n = idx >> 7, k = idx & 127;
    int l = mat >> 1, which = mat & 1;
    const float* W = which ? lin2_w : lin1_w;
    wt[gid] = (half_t)W[(size_t)l * DIMD * DIMD + k * DIMD + n];
}

// z0[dst] += h[src] for each edge; 32 threads x 4 floats per edge
__global__ void edge_scatter(const int* __restrict__ ei, const float* __restrict__ h,
                             float* __restrict__ z0, int E) {
    long gid = (long)blockIdx.x * blockDim.x + threadIdx.x;
    if (gid >= (long)E * 32) return;
    int e  = (int)(gid >> 5);
    int f0 = (int)(gid & 31) * 4;
    int s = ei[e];
    int d = ei[E + e];
    float4 v = *(const float4*)(h + (size_t)s * DIMD + f0);
    float* p = z0 + (size_t)d * DIMD + f0;
    atomicAdd(p + 0, v.x);
    atomicAdd(p + 1, v.y);
    atomicAdd(p + 2, v.z);
    atomicAdd(p + 3, v.w);
}

#define STAT_ROWS 512
__global__ void bn_stats(const float* __restrict__ t, int nrows,
                         float* __restrict__ sum, float* __restrict__ sumsq) {
    __shared__ float s1[256], s2[256];
    int ch  = threadIdx.x & 127;
    int hlf = threadIdx.x >> 7;
    int rend = blockIdx.x * STAT_ROWS + STAT_ROWS;
    if (rend > nrows) rend = nrows;
    float a = 0.f, b = 0.f;
    for (int r = blockIdx.x * STAT_ROWS + hlf; r < rend; r += 2) {
        float v = t[(size_t)r * DIMD + ch];
        a += v;
        b += v * v;
    }
    s1[threadIdx.x] = a;
    s2[threadIdx.x] = b;
    __syncthreads();
    if (threadIdx.x < 128) {
        atomicAdd(&sum[ch],   s1[threadIdx.x] + s1[threadIdx.x + 128]);
        atomicAdd(&sumsq[ch], s2[threadIdx.x] + s2[threadIdx.x + 128]);
    }
}

__global__ void bn_finalize(const float* __restrict__ sum, const float* __restrict__ sumsq,
                            const float* __restrict__ g, const float* __restrict__ b,
                            float* __restrict__ scale, float* __restrict__ shift, int nrows) {
    int c = threadIdx.x;
    if (c < DIMD) {
        float inv = 1.f / (float)nrows;
        float mu  = sum[c] * inv;
        float var = sumsq[c] * inv - mu * mu;
        float s   = g[c] * rsqrtf(var + 1e-5f);
        scale[c] = s;
        shift[c] = b[c] - mu * s;
    }
}

__global__ void pool_kernel(const float* __restrict__ h, const int* __restrict__ batch,
                            float* __restrict__ pooled, float* __restrict__ counts, int nrows) {
    long gid = (long)blockIdx.x * blockDim.x + threadIdx.x;
    if (gid >= (long)nrows * DIMD) return;
    int n = (int)(gid >> 7);
    int c = (int)(gid & 127);
    int g = batch[n];
    atomicAdd(&pooled[(size_t)g * DIMD + c], h[gid]);
    if (c == 0) atomicAdd(&counts[g], 1.f);
}

// readout MLP: 128 -> 64 -> 32 -> 16 ; one block per graph
__global__ void readout(const float* __restrict__ pooled, const float* __restrict__ counts,
                        const float* __restrict__ w1, const float* __restrict__ b1,
                        const float* __restrict__ w2, const float* __restrict__ b2,
                        const float* __restrict__ w3, const float* __restrict__ b3,
                        float* __restrict__ out) {
    __shared__ float p[128], h1[64], h2[32];
    int g = blockIdx.x;
    int t = threadIdx.x;
    float inv = 1.f / fmaxf(counts[g], 1.f);
    if (t < 128) p[t] = pooled[(size_t)g * DIMD + t] * inv;
    __syncthreads();
    if (t < 64) {
        float acc = b1[t];
        for (int k = 0; k < 128; ++k) acc += p[k] * w1[k * 64 + t];
        h1[t] = fmaxf(acc, 0.f);
    }
    __syncthreads();
    if (t < 32) {
        float acc = b2[t];
        for (int k = 0; k < 64; ++k) acc += h1[k] * w2[k * 32 + t];
        h2[t] = fmaxf(acc, 0.f);
    }
    __syncthreads();
    if (t < 16) {
        float acc = b3[t];
        for (int k = 0; k < 32; ++k) acc += h2[k] * w3[k * 16 + t];
        out[g * 16 + t] = acc;
    }
}

// ======================= launch ==========================
extern "C" void kernel_launch(void* const* d_in, const int* in_sizes, int n_in,
                              void* d_out, int out_size, void* d_ws, size_t ws_size,
                              hipStream_t stream) {
    const float* x      = (const float*)d_in[0];
    const int*   ei     = (const int*)d_in[1];
    // d_in[2] = edge_attr (unused by reference math)
    const int*   batch  = (const int*)d_in[3];
    const float* lin1_w = (const float*)d_in[4];
    const float* lin1_b = (const float*)d_in[5];
    const float* bn_g   = (const float*)d_in[6];
    const float* bn_b   = (const float*)d_in[7];
    const float* lin2_w = (const float*)d_in[8];
    const float* lin2_b = (const float*)d_in[9];
    const float* mlp_w1 = (const float*)d_in[10];
    const float* mlp_b1 = (const float*)d_in[11];
    const float* mlp_w2 = (const float*)d_in[12];
    const float* mlp_b2 = (const float*)d_in[13];
    const float* mlp_w3 = (const float*)d_in[14];
    const float* mlp_b3 = (const float*)d_in[15];
    float* out = (float*)d_out;

    const int N = in_sizes[0] / DIMD;
    const int E = in_sizes[1] / 2;
    const int G = out_size / 16;

    // workspace layout (two big N x 128 f32 ping-pong buffers + small scratch)
    size_t ND = (size_t)N * DIMD;
    float*  bufA  = (float*)d_ws;
    float*  bufB  = bufA + ND;
    half_t* wt    = (half_t*)(bufB + ND);           // 6 * 128*128 halves
    float*  stats = (float*)(wt + 6 * DIMD * DIMD);
    float*  sum   = stats;
    float*  sumsq = stats + 128;
    float*  scl   = stats + 256;
    float*  shf   = stats + 384;
    float*  pooled = stats + 512;                   // G*128
    float*  counts = pooled + (size_t)G * DIMD;     // G

    convert_weights<<<(6 * DIMD * DIMD + 255) / 256, 256, 0, stream>>>(lin1_w, lin2_w, wt);

    for (int l = 0; l < 3; ++l) {
        const float* h  = (l == 0) ? x : ((l == 1) ? bufB : bufA);
        float* z0 = (l % 2 == 0) ? bufB : bufA;   // also becomes layer output buffer
        float* t  = (l % 2 == 0) ? bufA : bufB;

        long n4 = (long)(ND / 4);
        copy_f4<<<(int)((n4 + 255) / 256), 256, 0, stream>>>((const float4*)h, (float4*)z0, n4);

        long ew = (long)E * 32;
        edge_scatter<<<(int)((ew + 255) / 256), 256, 0, stream>>>(ei, h, z0, E);

        gin_gemm<0, 0><<<(N + 127) / 128, 256, 0, stream>>>(
            z0, wt + (size_t)(l * 2) * DIMD * DIMD, lin1_b + l * DIMD, t, N,
            nullptr, nullptr);

        fill_zero<<<1, 256, 0, stream>>>(sum, 256);
        bn_stats<<<(N + STAT_ROWS - 1) / STAT_ROWS, 256, 0, stream>>>(t, N, sum, sumsq);
        bn_finalize<<<1, 128, 0, stream>>>(sum, sumsq, bn_g + l * DIMD, bn_b + l * DIMD,
                                           scl, shf, N);

        if (l < 2) {
            gin_gemm<1, 1><<<(N + 127) / 128, 256, 0, stream>>>(
                t, wt + (size_t)(l * 2 + 1) * DIMD * DIMD, lin2_b + l * DIMD, z0, N, scl, shf);
        } else {
            gin_gemm<1, 0><<<(N + 127) / 128, 256, 0, stream>>>(
                t, wt + (size_t)(l * 2 + 1) * DIMD * DIMD, lin2_b + l * DIMD, z0, N, scl, shf);
        }
    }

    float* hfin = bufB;   // layer-2 output lives in bufB
    fill_zero<<<(G * DIMD + G + 255) / 256, 256, 0, stream>>>(pooled, G * DIMD + G);
    long pw = (long)N * DIMD;
    pool_kernel<<<(int)((pw + 255) / 256), 256, 0, stream>>>(hfin, batch, pooled, counts, N);
    readout<<<G, 128, 0, stream>>>(pooled, counts, mlp_w1, mlp_b1, mlp_w2, mlp_b2,
                                   mlp_w3, mlp_b3, out);
}
